// TinyNet_38311108281050
// MI455X (gfx1250) — matrix operand order repaired and checked by reading, not compiled, not run on hardware
//
#include <hip/hip_runtime.h>
#include <hip/hip_bf16.h>

typedef __attribute__((ext_vector_type(16))) _Float16 v16h;
typedef __attribute__((ext_vector_type(8)))  _Float16 v8h;
typedef __attribute__((ext_vector_type(8)))  float    v8f;

constexpr int MM = 32;
constexpr int KK = 4096;
constexpr int NN = 12288;
constexpr int KSPLIT = 4;
constexpr int KCHUNK = KK / KSPLIT;        // 1024
constexpr int ASTRIDE = KCHUNK + 8;        // +8 halves (16B) pad: row stride = 516 dwords
                                           // => 4-bank skew per row, conflict-free ds_load_b128

// ---------------------------------------------------------------- zero d_out
__global__ __launch_bounds__(256) void qmm_zero(float* __restrict__ out, int count) {
    int i = blockIdx.x * 256 + threadIdx.x;
    if (i < count) out[i] = 0.0f;
}

// -------------------------------------------- prep: x -> f16, row sums of x
__global__ __launch_bounds__(256) void qmm_prep(const float* __restrict__ x,
                                                _Float16* __restrict__ xh,
                                                float* __restrict__ rowsum) {
    __shared__ float red[256];
    const int row = blockIdx.x;                 // 0..31
    const float* xr  = x  + (size_t)row * KK;
    _Float16*    xhr = xh + (size_t)row * KK;
    float local = 0.0f;
    for (int i = threadIdx.x; i < KK; i += 256) {
        float v = xr[i];
        xhr[i] = (_Float16)v;
        local += v;
    }
    red[threadIdx.x] = local;
    __syncthreads();
    for (int s = 128; s > 0; s >>= 1) {
        if ((int)threadIdx.x < s) red[threadIdx.x] += red[threadIdx.x + s];
        __syncthreads();
    }
    if (threadIdx.x == 0) rowsum[row] = red[0];
}

// ------------------------------------------------------------- main WMMA GEMM
// grid = (NN/128, KSPLIT), block = 256 (8 waves). Wave w owns n-tile
// [blockIdx.x*128 + w*16, +16) and both M halves (rows 0..15 and 16..31).
// A (x_h chunk, shared by all 8 waves) is staged once in LDS.
__global__ __launch_bounds__(256) void qmm_main(const int*      __restrict__ w,
                                                const _Float16* __restrict__ xh,
                                                const float*    __restrict__ rowsum,
                                                const _Float16* __restrict__ scale,
                                                const _Float16* __restrict__ zp,
                                                float*          __restrict__ out) {
    __shared__ _Float16 atile[MM * ASTRIDE];   // 32 x (1024+8) halves ~ 64.5 KB

    const int lane  = threadIdx.x & 31;
    const int wave  = threadIdx.x >> 5;
    const int nlane = lane & 15;
    const int n     = blockIdx.x * 128 + wave * 16 + nlane;
    const int kstart = blockIdx.y * KCHUNK;

    // ---- stage A chunk: rows 0..31, cols kstart..kstart+KCHUNK, 16B per copy
    for (int idx = threadIdx.x; idx < MM * (KCHUNK / 8); idx += 256) {
        const int r = idx >> 7;                // idx / (KCHUNK/8), KCHUNK/8 = 128
        const int c = (idx & 127) * 8;
        *(v8h*)(atile + r * ASTRIDE + c) =
            *(const v8h*)(xh + (size_t)r * KK + kstart + c);
    }
    __syncthreads();

    // A-fragment K offsets (contiguous, matching B):
    //   lanes 0-15:  elems 0..15 -> K = k0+0..15
    //   lanes 16-31: elems 0..15 -> K = k0+16..31
    const int abase = (lane < 16) ? 0 : 16;
    // B-fragment (16-bit B 32x16): lane column = lane%16,
    //   lanes 0-15 hold K = k0+0..15, lanes 16-31 hold K = k0+16..31
    const int kbase = (lane < 16) ? 0 : 16;

    const _Float16* ar0 = atile + nlane * ASTRIDE;          // A rows 0..15
    const _Float16* ar1 = atile + (nlane + 16) * ASTRIDE;   // A rows 16..31

    v8f acc0 = {};   // M = 0..15
    v8f acc1 = {};   // M = 16..31

    for (int kk = 0; kk < KCHUNK; kk += 32) {
        // ---- A fragments from LDS: two contiguous 16B ds loads per half
        v16h a0, a1;
        {
            const v8h lo0 = *(const v8h*)(ar0 + kk + abase);
            const v8h hi0 = *(const v8h*)(ar0 + kk + 8 + abase);
            const v8h lo1 = *(const v8h*)(ar1 + kk + abase);
            const v8h hi1 = *(const v8h*)(ar1 + kk + 8 + abase);
#pragma unroll
            for (int i = 0; i < 8; ++i) {
                a0[i] = lo0[i]; a0[8 + i] = hi0[i];
                a1[i] = lo1[i]; a1[8 + i] = hi1[i];
            }
        }
        // ---- B fragment: 16 K-rows of the int32 weight, NT (stream once)
        v16h b;
        const int* wp = w + (size_t)(kstart + kk + kbase) * NN + n;
#pragma unroll
        for (int t = 0; t < 16; ++t) {
            int wv = __builtin_nontemporal_load(wp);
            b[t] = (_Float16)(unsigned short)(unsigned)wv;  // 0..255 exact in f16
            wp += NN;
        }
        // ---- D = A*B + C  (f32 accumulate)
        acc0 = __builtin_amdgcn_wmma_f32_16x16x32_f16(false, a0, false, b,
                                                      (short)0, acc0, false, false);
        acc1 = __builtin_amdgcn_wmma_f32_16x16x32_f16(false, a1, false, b,
                                                      (short)0, acc1, false, false);
    }

    // Branchless epilogue: out[m,n] += s*S - corr*rowsum[m], corr = s*z only for
    // the blockIdx.y==0 partial (applied exactly once). f32 atomics combine splits.
    const float s    = (float)scale[n];
    const float corr = (blockIdx.y == 0) ? s * (float)zp[n] : 0.0f;
#pragma unroll
    for (int v = 0; v < 8; ++v) {
        // C/D layout: VGPR v -> M = v (lanes 0-15) or v+8 (lanes 16-31)
        const int m0 = (lane < 16) ? v : (v + 8);
        const float out0 = s * acc0[v] - corr * rowsum[m0];
        const float out1 = s * acc1[v] - corr * rowsum[m0 + 16];
        atomicAdd(out + (size_t)m0 * NN + n, out0);
        atomicAdd(out + (size_t)(m0 + 16) * NN + n, out1);
    }
}

extern "C" void kernel_launch(void* const* d_in, const int* in_sizes, int n_in,
                              void* d_out, int out_size, void* d_ws, size_t ws_size,
                              hipStream_t stream) {
    const float*    x     = (const float*)d_in[0];
    const int*      w     = (const int*)d_in[1];
    const _Float16* scale = (const _Float16*)d_in[2];
    const _Float16* zp    = (const _Float16*)d_in[3];
    float* out = (float*)d_out;

    _Float16* xh     = (_Float16*)d_ws;                                   // 256 KB
    float*    rowsum = (float*)((char*)d_ws + (size_t)MM * KK * sizeof(_Float16));

    qmm_zero<<<(MM * NN + 255) / 256, 256, 0, stream>>>(out, MM * NN);
    qmm_prep<<<MM, 256, 0, stream>>>(x, xh, rowsum);
    qmm_main<<<dim3(NN / 128, KSPLIT), 256, 0, stream>>>(w, xh, rowsum, scale, zp, out);
}
